// Attention_74663711474318
// MI455X (gfx1250) — matrix operand order repaired and checked by reading, not compiled, not run on hardware
//
#include <hip/hip_runtime.h>
#include <hip/hip_bf16.h>

// ---------------------------------------------------------------------------
// MI455X (gfx1250) fused multi-head attention, compute-bound -> bf16 WMMA.
//   SEQ=4096, HIDDEN=4096, HEAD_DIM=1024, NUM_HEAD=4
// Pipeline:
//   1) fp32 -> bf16 conversion of h; fp32 -> bf16 + transpose of Wq/Wk/Wv
//   2) generic 128x128x32-tile bf16 WMMA GEMM (f32 accum), staged through LDS
//      with GLOBAL_LOAD_ASYNC_TO_LDS_B128 (ASYNCcnt-tracked, triple-buffered,
//      one barrier per K-step) when the toolchain exposes the builtin.
//   3) per head: scores = q @ k^T * 1/32, f32 row softmax -> bf16 P,
//      out = P @ V^T written into d_out columns.
// ---------------------------------------------------------------------------

typedef __bf16 bf16;
typedef __attribute__((ext_vector_type(4)))  bf16  bf16x4;
typedef __attribute__((ext_vector_type(8)))  bf16  bf16x8;
typedef __attribute__((ext_vector_type(16))) bf16  bf16x16;
typedef __attribute__((ext_vector_type(8)))  float f32x8;
typedef __attribute__((ext_vector_type(4)))  int   v4i;

static constexpr int SEQc = 4096;
static constexpr int HIDc = 4096;
static constexpr int Ec   = 1024;
static constexpr int NHc  = 4;

#if __has_builtin(__builtin_amdgcn_global_load_async_to_lds_b128) && \
    __has_builtin(__builtin_amdgcn_s_wait_asynccnt)
#define HAS_ASYNC_LDS 1
#else
#define HAS_ASYNC_LDS 0
#endif

#if HAS_ASYNC_LDS
// Copy 16B/lane (512B per wave-instruction) global -> LDS via the CDNA5
// async path (tracked by ASYNCcnt, no VGPR round trip). Builtin signature
// (probe-confirmed via diagnostics): (v4i* global_src, v4i* lds_dst, imm, imm)
// with generic pointers; the backend infers the address spaces.
__device__ __forceinline__ void async_cp16(const bf16* g, bf16* l) {
  __builtin_amdgcn_global_load_async_to_lds_b128((v4i*)g, (v4i*)l, 0, 0);
}
#endif

// ---------------- fp32 -> bf16 elementwise ----------------
__global__ __launch_bounds__(256)
void f32_to_bf16_kernel(const float* __restrict__ in, bf16* __restrict__ out, int n) {
  int i = (blockIdx.x * 256 + threadIdx.x) * 4;
  if (i + 3 < n) {
    float4 v = *(const float4*)(in + i);
    bf16x4 o;
    o[0] = (bf16)v.x; o[1] = (bf16)v.y; o[2] = (bf16)v.z; o[3] = (bf16)v.w;
    *(bf16x4*)(out + i) = o;
  }
}

// ---------------- fp32 [R][C] -> bf16 [C][R] (per blockIdx.z slice) ----------------
__global__ __launch_bounds__(256)
void f32_to_bf16_transpose_kernel(const float* __restrict__ in, bf16* __restrict__ out,
                                  int R, int C) {
  __shared__ float tile[32][33];
  const float* inz  = in  + (size_t)blockIdx.z * R * C;
  bf16*        outz = out + (size_t)blockIdx.z * R * C;
  const int c0 = blockIdx.x * 32, r0 = blockIdx.y * 32;
  #pragma unroll
  for (int i = 0; i < 32; i += 8)
    tile[threadIdx.y + i][threadIdx.x] =
        inz[(size_t)(r0 + threadIdx.y + i) * C + (c0 + threadIdx.x)];
  __syncthreads();
  #pragma unroll
  for (int i = 0; i < 32; i += 8)
    outz[(size_t)(c0 + threadIdx.y + i) * R + (r0 + threadIdx.x)] =
        (bf16)tile[threadIdx.x][threadIdx.y + i];
}

// ---------------- generic bf16 WMMA GEMM ----------------
// C[m,n] = alpha * sum_k A[m,k]*B[n,k] + bias[n]
//   A: row-major [M][K] bf16 (lda), B: [N][K] bf16 (ldb)  (B already "transposed")
//   C element (m,n) stored at  C + z*sCz + m*outRS + n*outCS   (f32 or bf16)
template <bool OUT_BF16>
__global__ __launch_bounds__(256)
void gemm_bf16_wmma(const bf16* __restrict__ A, long long lda, long long sAz,
                    const bf16* __restrict__ B, long long ldb, long long sBz,
                    const float* __restrict__ bias, long long sBiasz,
                    void* __restrict__ C, long long outRS, long long outCS, long long sCz,
                    int M, int N, int K, float alpha) {
  constexpr int BM = 128, BN = 128, BK = 32, LDT = BK + 8;  // stride 40 elems (80B)
#if HAS_ASYNC_LDS
  constexpr int NBUF = 3;  // triple buffer: WAR-safe with ONE barrier per K-step
#else
  constexpr int NBUF = 1;
#endif
  __shared__ bf16 ldsA[NBUF * BM * LDT];
  __shared__ bf16 ldsB[NBUF * BN * LDT];

  A += (size_t)blockIdx.z * sAz;
  B += (size_t)blockIdx.z * sBz;
  if (bias) bias += (size_t)blockIdx.z * sBiasz;

  const int tid  = threadIdx.x;
  const int lane = tid & 31;
  const int wave = tid >> 5;          // 8 waves
  const int l15  = lane & 15;
  const int half = (lane >> 4) & 1;   // wave32 half
  const int wm   = wave & 3;          // 4 waves along M
  const int wn   = wave >> 2;         // 2 waves along N
  const int blockM = blockIdx.y * BM;
  const int blockN = blockIdx.x * BN;

  f32x8 acc[2][4];
  #pragma unroll
  for (int i = 0; i < 2; ++i)
    #pragma unroll
    for (int j = 0; j < 4; ++j)
      #pragma unroll
      for (int e = 0; e < 8; ++e) acc[i][j][e] = 0.0f;

  const int NIT = K / BK;

#if HAS_ASYNC_LDS
  // 16B chunks: tile is 128 rows x 64B -> 512 chunks; each thread issues 2 per
  // operand => 4 async instructions per wave per stage.
  auto issueStage = [&](int kk, int buf) {
    bf16* dA = ldsA + buf * (BM * LDT);
    bf16* dB = ldsB + buf * (BN * LDT);
    #pragma unroll
    for (int i = 0; i < 2; ++i) {
      const int c   = tid + i * 256;       // 0..511
      const int row = c >> 2;              // 0..127
      const int col = (c & 3) * 8;         // elem offset, 16B granules
      async_cp16(A + (size_t)(blockM + row) * lda + kk + col, dA + row * LDT + col);
      async_cp16(B + (size_t)(blockN + row) * ldb + kk + col, dB + row * LDT + col);
    }
  };

  issueStage(0, 0);
  for (int it = 0; it < NIT; ++it) {
    const int nextk = (it + 1 < NIT) ? (it + 1) * BK : 0;  // tail: redundant refetch keeps counts uniform
    issueStage(nextk, (it + 1) % NBUF);
    // 8 outstanding now; oldest 4 (this stage) retire in order -> wait to <=4.
    __builtin_amdgcn_s_wait_asynccnt(4);
    __syncthreads();

    const bf16* bufA = ldsA + (it % NBUF) * (BM * LDT);
    const bf16* bufB = ldsB + (it % NBUF) * (BN * LDT);
#else
  for (int it = 0; it < NIT; ++it) {
    const int kk = it * BK;
    {  // stage A tile synchronously
      const int sRow = tid >> 1, sCh = (tid & 1) * 16;
      const bf16* src = A + (size_t)(blockM + sRow) * lda + kk + sCh;
      *(bf16x8*)(ldsA + sRow * LDT + sCh)     = *(const bf16x8*)(src);
      *(bf16x8*)(ldsA + sRow * LDT + sCh + 8) = *(const bf16x8*)(src + 8);
      const bf16* srcB = B + (size_t)(blockN + sRow) * ldb + kk + sCh;
      *(bf16x8*)(ldsB + sRow * LDT + sCh)     = *(const bf16x8*)(srcB);
      *(bf16x8*)(ldsB + sRow * LDT + sCh + 8) = *(const bf16x8*)(srcB + 8);
    }
    __syncthreads();
    const bf16* bufA = ldsA;
    const bf16* bufB = ldsB;
#endif

    // A fragment (16x32 bf16): lanes 0-15: M=lane, K runs {0..7,16..23};
    //                          lanes 16-31: M=lane-16, K runs {8..15,24..31}
    bf16x16 afrag[2];
    #pragma unroll
    for (int i = 0; i < 2; ++i) {
      const bf16* p = bufA + (wm * 32 + i * 16 + l15) * LDT + half * 8;
      bf16x8 lo = *(const bf16x8*)(p);
      bf16x8 hi = *(const bf16x8*)(p + 16);
      afrag[i] = __builtin_shufflevector(lo, hi, 0, 1, 2, 3, 4, 5, 6, 7,
                                                 8, 9, 10, 11, 12, 13, 14, 15);
    }
    // B fragment (32x16 bf16): lane holds column N=l15, K runs {0..7,16..23} / {8..15,24..31}
    bf16x16 bfrag[4];
    #pragma unroll
    for (int j = 0; j < 4; ++j) {
      const bf16* p = bufB + (wn * 64 + j * 16 + l15) * LDT + half * 8;
      bf16x8 lo = *(const bf16x8*)(p);
      bf16x8 hi = *(const bf16x8*)(p + 16);
      bfrag[j] = __builtin_shufflevector(lo, hi, 0, 1, 2, 3, 4, 5, 6, 7,
                                                 8, 9, 10, 11, 12, 13, 14, 15);
    }

    #pragma unroll
    for (int i = 0; i < 2; ++i)
      #pragma unroll
      for (int j = 0; j < 4; ++j)
        acc[i][j] = __builtin_amdgcn_wmma_f32_16x16x32_bf16(
            false, afrag[i], false, bfrag[j], (short)0, acc[i][j], false, false);

#if !HAS_ASYNC_LDS
    __syncthreads();  // single-buffer WAR barrier
#endif
  }

  // epilogue: C/D layout: VGPR r -> M = r + 8*half, N = l15
  #pragma unroll
  for (int i = 0; i < 2; ++i) {
    #pragma unroll
    for (int j = 0; j < 4; ++j) {
      const int n = blockN + wn * 64 + j * 16 + l15;
      const float bn = bias ? bias[n] : 0.0f;
      #pragma unroll
      for (int r = 0; r < 8; ++r) {
        const int m = blockM + wm * 32 + i * 16 + half * 8 + r;
        const float val = alpha * acc[i][j][r] + bn;
        const size_t off = (size_t)blockIdx.z * sCz + (size_t)m * outRS + (size_t)n * outCS;
        if (OUT_BF16) ((bf16*)C)[off]  = (bf16)val;
        else          ((float*)C)[off] = val;
      }
    }
  }
}

// ---------------- row softmax: f32 [rows][4096] -> bf16 ----------------
__global__ __launch_bounds__(256)
void softmax_rows_kernel(const float* __restrict__ S, bf16* __restrict__ P) {
  const int row = blockIdx.x;
  const float* s = S + (size_t)row * 4096;
  bf16* p = P + (size_t)row * 4096;
  const int tid = threadIdx.x;
  __shared__ float red[8];

  float v[16];
  float mx = -3.402823466e38f;
  #pragma unroll
  for (int i = 0; i < 16; ++i) {
    v[i] = s[tid + i * 256];
    mx = fmaxf(mx, v[i]);
  }
  for (int o = 16; o > 0; o >>= 1) mx = fmaxf(mx, __shfl_xor(mx, o, 32));
  if ((tid & 31) == 0) red[tid >> 5] = mx;
  __syncthreads();
  mx = red[0];
  #pragma unroll
  for (int w = 1; w < 8; ++w) mx = fmaxf(mx, red[w]);

  float sum = 0.0f;
  #pragma unroll
  for (int i = 0; i < 16; ++i) {
    v[i] = __expf(v[i] - mx);
    sum += v[i];
  }
  for (int o = 16; o > 0; o >>= 1) sum += __shfl_xor(sum, o, 32);
  __syncthreads();
  if ((tid & 31) == 0) red[tid >> 5] = sum;
  __syncthreads();
  sum = 0.0f;
  #pragma unroll
  for (int w = 0; w < 8; ++w) sum += red[w];
  const float inv = 1.0f / sum;
  #pragma unroll
  for (int i = 0; i < 16; ++i) p[tid + i * 256] = (bf16)(v[i] * inv);
}

// ---------------------------------------------------------------------------
extern "C" void kernel_launch(void* const* d_in, const int* in_sizes, int n_in,
                              void* d_out, int out_size, void* d_ws, size_t ws_size,
                              hipStream_t stream) {
  (void)in_sizes; (void)n_in; (void)out_size; (void)ws_size;

  const float* h  = (const float*)d_in[0];
  const float* Wq = (const float*)d_in[1];
  const float* bq = (const float*)d_in[2];
  const float* Wk = (const float*)d_in[3];
  const float* bk = (const float*)d_in[4];
  const float* Wv = (const float*)d_in[5];
  const float* bv = (const float*)d_in[6];
  float* out = (float*)d_out;

  // workspace carve-out (~320 MB total)
  char* ws = (char*)d_ws;
  size_t off = 0;
  auto carve = [&](size_t bytes) {
    void* p = ws + off;
    off += (bytes + 255) & ~(size_t)255;
    return p;
  };
  bf16*  hB  = (bf16*)carve((size_t)SEQc * HIDc * 2);        // h in bf16
  bf16*  WqT = (bf16*)carve((size_t)NHc * HIDc * Ec * 2);    // [H][E][D]
  bf16*  WkT = (bf16*)carve((size_t)NHc * HIDc * Ec * 2);
  bf16*  WvT = (bf16*)carve((size_t)NHc * HIDc * Ec * 2);
  bf16*  qB  = (bf16*)carve((size_t)NHc * SEQc * Ec * 2);    // [H][S][E]
  bf16*  kB  = (bf16*)carve((size_t)NHc * SEQc * Ec * 2);    // [H][S][E]
  bf16*  vT  = (bf16*)carve((size_t)NHc * Ec * SEQc * 2);    // [H][E][S]
  float* Sb  = (float*)carve((size_t)SEQc * SEQc * 4);       // one head of scores
  bf16*  Pb  = (bf16*)carve((size_t)SEQc * SEQc * 2);        // one head of softmax(P)

  // 1) conversions
  f32_to_bf16_kernel<<<(SEQc * HIDc) / 1024, 256, 0, stream>>>(h, hB, SEQc * HIDc);
  dim3 tb(32, 8);
  dim3 tg(Ec / 32, HIDc / 32, NHc);
  f32_to_bf16_transpose_kernel<<<tg, tb, 0, stream>>>(Wq, WqT, HIDc, Ec);
  f32_to_bf16_transpose_kernel<<<tg, tb, 0, stream>>>(Wk, WkT, HIDc, Ec);
  f32_to_bf16_transpose_kernel<<<tg, tb, 0, stream>>>(Wv, WvT, HIDc, Ec);

  // 2) projections: [S,D]x[D,E]+b per head (grid.z = heads)
  dim3 gp(Ec / 128, SEQc / 128, NHc);
  gemm_bf16_wmma<true><<<gp, 256, 0, stream>>>(
      hB, HIDc, 0, WqT, HIDc, (long long)HIDc * Ec, bq, Ec,
      qB, Ec, 1, (long long)SEQc * Ec, SEQc, Ec, HIDc, 1.0f);
  gemm_bf16_wmma<true><<<gp, 256, 0, stream>>>(
      hB, HIDc, 0, WkT, HIDc, (long long)HIDc * Ec, bk, Ec,
      kB, Ec, 1, (long long)SEQc * Ec, SEQc, Ec, HIDc, 1.0f);
  // v stored transposed -> vT[h][e][t] so it is directly the B operand of P@V
  gemm_bf16_wmma<true><<<gp, 256, 0, stream>>>(
      hB, HIDc, 0, WvT, HIDc, (long long)HIDc * Ec, bv, Ec,
      vT, 1, SEQc, (long long)Ec * SEQc, SEQc, Ec, HIDc, 1.0f);

  // 3) attention per head (reuses one S/P scratch pair)
  for (int hh = 0; hh < NHc; ++hh) {
    const bf16* qh = qB + (size_t)hh * SEQc * Ec;
    const bf16* kh = kB + (size_t)hh * SEQc * Ec;   // [t][e] == [N][K] for Q@K^T
    const bf16* vh = vT + (size_t)hh * Ec * SEQc;   // [e][t] == [N][K] for P@V

    dim3 gs(SEQc / 128, SEQc / 128, 1);
    gemm_bf16_wmma<false><<<gs, 256, 0, stream>>>(
        qh, Ec, 0, kh, Ec, 0, nullptr, 0,
        Sb, SEQc, 1, 0, SEQc, SEQc, Ec, 0.03125f /* 1/sqrt(1024) */);

    softmax_rows_kernel<<<SEQc, 256, 0, stream>>>(Sb, Pb);

    dim3 go(Ec / 128, SEQc / 128, 1);
    gemm_bf16_wmma<false><<<go, 256, 0, stream>>>(
        Pb, SEQc, 0, vh, SEQc, 0, nullptr, 0,
        out + (size_t)hh * Ec, (long long)NHc * Ec, 1, 0, SEQc, Ec, SEQc, 1.0f);
  }
}
